// MultiHeadAttention_60971355734594
// MI455X (gfx1250) — compile-verified, matrix operand order checked by reading, not run on hardware
//
#include <hip/hip_runtime.h>
#include <hip/hip_bf16.h>

// Multi-head attention forward for MI455X (gfx1250, wave32, WMMA).
// Pipeline: [GEMM x@Wqkv+b -> bf16] -> [flash attention, bf16 WMMA, fp32 softmax]
//           -> [GEMM vals@Wout+b -> fp32 out]
// All matrix math uses v_wmma_f32_16x16x32_bf16 (fp32 accumulate).

#define EMBED   1024
#define HEADS   16
#define HD      64
#define BATCH   2
#define SEQ     2048
#define MROWS   (BATCH * SEQ)   // 4096
#define QKV_N   (3 * EMBED)     // 3072
#define NEG_INF_F (-1.0e9f)
#define SM_SCALE  (0.125f)      // 1/sqrt(64)

typedef __attribute__((ext_vector_type(16))) __bf16 v16bf;
typedef __attribute__((ext_vector_type(8)))  float  v8f;
typedef __attribute__((ext_vector_type(4)))  float  f32x4;

union BF16Frag { v16bf v; f32x4 q[2]; };

// fp32 -> bf16 (round to nearest even), pure integer path.
static __device__ inline unsigned short f2bf(float f) {
  unsigned int x = __float_as_uint(f);
  unsigned int r = (x + 0x7FFFu + ((x >> 16) & 1u)) >> 16;
  return (unsigned short)r;
}

static __device__ inline v8f wmma_bf16(const BF16Frag& a, const BF16Frag& b, v8f c) {
  // (neg_a, A, neg_b, B, c_mod, C, reuse_a, reuse_b)
  return __builtin_amdgcn_wmma_f32_16x16x32_bf16(false, a.v, false, b.v,
                                                 (short)0, c, false, false);
}

// A-operand loader overloads (fp32 global -> bf16, or pass-through bf16).
static __device__ inline unsigned short load_abf(const float* A, int idx) { return f2bf(A[idx]); }
static __device__ inline unsigned short load_abf(const unsigned short* A, int idx) { return A[idx]; }
// C-store overloads (fp32 or bf16 destination).
static __device__ inline void store_c(float* C, int idx, float v) { C[idx] = v; }
static __device__ inline void store_c(unsigned short* C, int idx, float v) { C[idx] = f2bf(v); }

// ---------------------------------------------------------------------------
// Tiled GEMM with bias: C[M,N] = A[M,K] @ B[K,N] + bias[N]
// Block tile 128x128, 256 threads = 8 waves (2 wave-rows x 4 wave-cols);
// each wave owns a 64x32 tile (4x2 grid of 16x16 fp32 accumulators) =
// 8 WMMA per K-step per wave, 64 per block-iteration between barriers.
// B is staged *transposed* in LDS so both A and B WMMA fragments are
// contiguous 16-byte ds_load_b128 matching the gfx1250 16-bit A/B layout
// (lanes 0-15: K=0..7,16..23 ; lanes 16-31: +8). Next K-slab is prefetched
// with global_prefetch_b8 while the current one is consumed.
// ---------------------------------------------------------------------------
template <typename AT, typename OT>
__global__ __launch_bounds__(256)
void gemm_bias_wmma(const AT* __restrict__ A, const float* __restrict__ Bw,
                    const float* __restrict__ bias, OT* __restrict__ C,
                    int M, int N, int K)
{
  __shared__ __align__(16) unsigned short As[128][40]; // [m][k] k-pad 32->40 (80B rows)
  __shared__ __align__(16) unsigned short Bt[128][40]; // [n][k] transposed

  const int tid  = threadIdx.x;
  const int lane = tid & 31;
  const int w    = tid >> 5;
  const int wr   = w >> 2;      // 2 wave-rows (64 rows each)
  const int wc   = w & 3;       // 4 wave-cols (32 cols each)
  const int l16  = lane & 15;
  const int half = lane >> 4;
  const int m0   = blockIdx.y * 128;
  const int n0   = blockIdx.x * 128;

  v8f zero = {};
  v8f acc[4][2];
  #pragma unroll
  for (int i = 0; i < 4; ++i) { acc[i][0] = zero; acc[i][1] = zero; }

  for (int k0 = 0; k0 < K; k0 += 32) {
    // Stage A tile 128x32 (convert to bf16 if needed).
    #pragma unroll
    for (int e = 0; e < 16; ++e) {
      int flat = tid + e * 256;
      int r = flat >> 5, c = flat & 31;
      As[r][c] = load_abf(A, (m0 + r) * K + (k0 + c));
    }
    // Stage B tile 32x128 transposed -> Bt[n][k].
    #pragma unroll
    for (int e = 0; e < 16; ++e) {
      int flat = tid + e * 256;
      int r = flat >> 7, c = flat & 127;          // r = k, c = n
      Bt[c][r] = f2bf(Bw[(k0 + r) * N + (n0 + c)]);
    }
    // Prefetch next K-slab while this one is consumed (global_prefetch_b8).
    if (k0 + 32 < K) {
      __builtin_prefetch(&A[(m0 + (tid >> 5)) * K + (k0 + 32) + (tid & 31)], 0, 1);
      __builtin_prefetch(&Bw[(k0 + 32 + (tid >> 7)) * N + n0 + (tid & 127)], 0, 1);
    }
    __syncthreads();

    BF16Frag af[4], bf[2];
    #pragma unroll
    for (int i = 0; i < 4; ++i) {
      int row = wr * 64 + i * 16 + l16;
      af[i].q[0] = *(const f32x4*)&As[row][half * 8];       // K = 0..7  (+8 hi-half)
      af[i].q[1] = *(const f32x4*)&As[row][16 + half * 8];  // K = 16..23 (+8 hi-half)
    }
    #pragma unroll
    for (int j = 0; j < 2; ++j) {
      int col = wc * 32 + j * 16 + l16;
      bf[j].q[0] = *(const f32x4*)&Bt[col][half * 8];
      bf[j].q[1] = *(const f32x4*)&Bt[col][16 + half * 8];
    }
    #pragma unroll
    for (int i = 0; i < 4; ++i)
      #pragma unroll
      for (int j = 0; j < 2; ++j)
        acc[i][j] = wmma_bf16(af[i], bf[j], acc[i][j]);
    __syncthreads();
  }

  // Epilogue: C-frag layout -> VGPR r holds rows (r, r+8) split across lane halves.
  #pragma unroll
  for (int i = 0; i < 4; ++i)
    #pragma unroll
    for (int j = 0; j < 2; ++j) {
      int col = n0 + wc * 32 + j * 16 + l16;
      float bv = bias[col];
      #pragma unroll
      for (int r = 0; r < 8; ++r) {
        int row = m0 + wr * 64 + i * 16 + r + 8 * half;
        store_c(C, row * N + col, acc[i][j][r] + bv);
      }
    }
}

// ---------------------------------------------------------------------------
// Flash attention (bf16 WMMA, fp32 online softmax).
// Grid: (SEQ/128, BATCH*HEADS), 256 threads = 8 waves, 16 queries per wave
// (128 per workgroup so K/V staging is amortized over 2x the compute).
// Per 32-key tile: K staged [key][d], V staged transposed [d][key]; scores via
// 4 WMMA, mask (key <= query gets NEG_INF, faithful to reference tril mask),
// half-wave shuffle reductions for row max/sum, P round-tripped through a
// per-wave LDS buffer to convert C-fragments into A-fragments, then 4 WMMA
// for P@V.
// ---------------------------------------------------------------------------
__global__ __launch_bounds__(256)
void flash_attn_wmma(const unsigned short* __restrict__ qkv,
                     unsigned short* __restrict__ vals)
{
  __shared__ __align__(16) unsigned short Ks[32][72];    // [key][d], pad 64->72
  __shared__ __align__(16) unsigned short Vt[64][40];    // [d][key], pad 32->40
  __shared__ __align__(16) unsigned short Pb[8][16][40]; // per-wave P staging

  const int tid  = threadIdx.x;
  const int lane = tid & 31;
  const int w    = tid >> 5;
  const int l16  = lane & 15;
  const int half = lane >> 4;
  const int bh   = blockIdx.y;
  const int b    = bh >> 4;
  const int h    = bh & 15;
  const int q0   = blockIdx.x * 128 + w * 16;

  // qkv row layout per head: [q(64) | k(64) | v(64)] at column h*192.
  const unsigned short* base = qkv + (size_t)(b * SEQ) * QKV_N + h * (3 * HD);

  // Load Q fragments once (2 frags cover hd=64 in K-dim).
  BF16Frag qa[2];
  {
    const unsigned short* qr = base + (size_t)(q0 + l16) * QKV_N;
    qa[0].q[0] = *(const f32x4*)(qr + half * 8);
    qa[0].q[1] = *(const f32x4*)(qr + 16 + half * 8);
    qa[1].q[0] = *(const f32x4*)(qr + 32 + half * 8);
    qa[1].q[1] = *(const f32x4*)(qr + 48 + half * 8);
  }

  v8f zero = {};
  v8f o[4]; o[0] = zero; o[1] = zero; o[2] = zero; o[3] = zero;
  float m[8], l[8];
  #pragma unroll
  for (int r = 0; r < 8; ++r) { m[r] = -3.0e38f; l[r] = 0.f; }

  for (int kt = 0; kt < SEQ / 32; ++kt) {
    const int key0 = kt * 32;
    const unsigned short* kv = base + (size_t)key0 * QKV_N;

    // Stage K (row-major) and V (transposed) tiles; data already bf16.
    #pragma unroll
    for (int e = 0; e < 8; ++e) {
      int flat = tid + e * 256;
      int kk = flat >> 6, d = flat & 63;
      const unsigned short* row = kv + (size_t)kk * QKV_N;
      Ks[kk][d] = row[HD + d];        // K
      Vt[d][kk] = row[2 * HD + d];    // V, transposed for P@V B-fragments
    }
    __syncthreads();

    // Scores S = Q @ K^T for 16q x 32k tile (2 C-frags, 4 WMMA).
    v8f sc0 = zero, sc1 = zero;
    {
      BF16Frag kb;
      int krow = l16;                                   // keys 0..15
      kb.q[0] = *(const f32x4*)&Ks[krow][half * 8];
      kb.q[1] = *(const f32x4*)&Ks[krow][16 + half * 8];
      sc0 = wmma_bf16(qa[0], kb, sc0);
      kb.q[0] = *(const f32x4*)&Ks[krow][32 + half * 8];
      kb.q[1] = *(const f32x4*)&Ks[krow][48 + half * 8];
      sc0 = wmma_bf16(qa[1], kb, sc0);
      krow = 16 + l16;                                  // keys 16..31
      kb.q[0] = *(const f32x4*)&Ks[krow][half * 8];
      kb.q[1] = *(const f32x4*)&Ks[krow][16 + half * 8];
      sc1 = wmma_bf16(qa[0], kb, sc1);
      kb.q[0] = *(const f32x4*)&Ks[krow][32 + half * 8];
      kb.q[1] = *(const f32x4*)&Ks[krow][48 + half * 8];
      sc1 = wmma_bf16(qa[1], kb, sc1);
    }

    // Online softmax per row (C layout: VGPR r <-> rows r / r+8 per lane-half;
    // xor shuffles with mask<16 stay within each 16-lane half = one row).
    #pragma unroll
    for (int r = 0; r < 8; ++r) {
      int qrow = q0 + r + 8 * half;
      float s0 = sc0[r] * SM_SCALE;
      float s1 = sc1[r] * SM_SCALE;
      if (key0 + l16      <= qrow) s0 += NEG_INF_F;   // reference masks k <= q
      if (key0 + 16 + l16 <= qrow) s1 += NEG_INF_F;
      float tmax = fmaxf(s0, s1);
      #pragma unroll
      for (int off = 1; off < 16; off <<= 1)
        tmax = fmaxf(tmax, __shfl_xor(tmax, off, 32));
      float mn    = fmaxf(m[r], tmax);
      float alpha = __expf(m[r] - mn);
      float p0 = __expf(s0 - mn);
      float p1 = __expf(s1 - mn);
      float ps = p0 + p1;
      #pragma unroll
      for (int off = 1; off < 16; off <<= 1)
        ps += __shfl_xor(ps, off, 32);
      l[r] = l[r] * alpha + ps;
      m[r] = mn;
      o[0][r] *= alpha; o[1][r] *= alpha; o[2][r] *= alpha; o[3][r] *= alpha;
      int prow = r + 8 * half;
      Pb[w][prow][l16]      = f2bf(p0);
      Pb[w][prow][16 + l16] = f2bf(p1);
    }
    // Per-wave LDS buffer: DS ops are in-order within a wave, no barrier needed.

    // O += P @ V  (P as one 16x32 A-frag, V as 4 B-frags over d).
    BF16Frag pa;
    pa.q[0] = *(const f32x4*)&Pb[w][l16][half * 8];
    pa.q[1] = *(const f32x4*)&Pb[w][l16][16 + half * 8];
    #pragma unroll
    for (int n = 0; n < 4; ++n) {
      BF16Frag vb;
      int vrow = n * 16 + l16;
      vb.q[0] = *(const f32x4*)&Vt[vrow][half * 8];
      vb.q[1] = *(const f32x4*)&Vt[vrow][16 + half * 8];
      o[n] = wmma_bf16(pa, vb, o[n]);
    }
    __syncthreads();
  }

  // Normalize and write vals[B*S, E] (head-major columns h*64+d), bf16.
  #pragma unroll
  for (int n = 0; n < 4; ++n) {
    int col = h * HD + n * 16 + l16;
    #pragma unroll
    for (int r = 0; r < 8; ++r) {
      int row = b * SEQ + q0 + r + 8 * half;
      vals[(size_t)row * EMBED + col] = f2bf(o[n][r] / l[r]);
    }
  }
}

// ---------------------------------------------------------------------------
extern "C" void kernel_launch(void* const* d_in, const int* in_sizes, int n_in,
                              void* d_out, int out_size, void* d_ws, size_t ws_size,
                              hipStream_t stream)
{
  (void)in_sizes; (void)n_in; (void)out_size; (void)ws_size;
  const float* x    = (const float*)d_in[0];
  const float* Wqkv = (const float*)d_in[1];
  const float* bqkv = (const float*)d_in[2];
  const float* Wout = (const float*)d_in[3];
  const float* bout = (const float*)d_in[4];
  float* out = (float*)d_out;

  // Scratch: qkv bf16 [4096 x 3072] (24 MB) + vals bf16 [4096 x 1024] (8 MB).
  unsigned short* qkv  = (unsigned short*)d_ws;
  unsigned short* vals = qkv + (size_t)MROWS * QKV_N;

  gemm_bias_wmma<float, unsigned short>
      <<<dim3(QKV_N / 128, MROWS / 128), 256, 0, stream>>>(
          x, Wqkv, bqkv, qkv, MROWS, QKV_N, EMBED);

  flash_attn_wmma<<<dim3(SEQ / 128, BATCH * HEADS), 256, 0, stream>>>(qkv, vals);

  gemm_bias_wmma<unsigned short, float>
      <<<dim3(EMBED / 128, MROWS / 128), 256, 0, stream>>>(
          vals, Wout, bout, out, MROWS, EMBED, EMBED);
}
